// MoETransformerLayer_21655225106533
// MI455X (gfx1250) — matrix operand order repaired and checked by reading, not compiled, not run on hardware
//
#include <hip/hip_runtime.h>
#include <hip/hip_bf16.h>
#include <math.h>

// ---------------- problem constants (from reference) ----------------
#define S_  2048
#define B_  4
#define E_  1024
#define H_  16
#define HD_ 64
#define FF_ 4096
#define NE_ 8
#define T_  (S_ * B_)     // 8192 tokens
#define E3_ (3 * E_)      // 3072

typedef _Float16 f16;
typedef __attribute__((ext_vector_type(16))) _Float16 v16h;
typedef __attribute__((ext_vector_type(8)))  _Float16 v8h;
typedef __attribute__((ext_vector_type(4)))  _Float16 v4h;
typedef __attribute__((ext_vector_type(8)))  float    v8f;
typedef __attribute__((ext_vector_type(4)))  unsigned int u32x4;
typedef __attribute__((ext_vector_type(8)))  unsigned int u32x8;

#define SHUF16(lo, hh) __builtin_shufflevector(lo, hh, 0,1,2,3,4,5,6,7,8,9,10,11,12,13,14,15)

enum { EPI_F32 = 0, EPI_QKV = 1, EPI_GELU = 2, EPI_RESID = 3, EPI_COMBINE = 4 };

// ---------------- fp32 -> fp16 convert (already [N,K] weights) ----------------
__global__ void cvt_f32_to_f16(const float* __restrict__ src, f16* __restrict__ dst, size_t n) {
    size_t i = (size_t)blockIdx.x * blockDim.x + threadIdx.x;
    size_t stride = (size_t)gridDim.x * blockDim.x;
    for (; i < n; i += stride) dst[i] = (f16)src[i];
}

// ---------------- batched fp32 -> fp16 transposing convert ----------------
// src: [Z][K][N] fp32 ; dst: [Z][N][K] fp16. 32x32 tiles through LDS.
__global__ __launch_bounds__(256) void cvt_tr_f32_to_f16(
    const float* __restrict__ src, f16* __restrict__ dst, int K, int N) {
    __shared__ float tile[32][33];
    size_t eoff = (size_t)blockIdx.z * K * N;
    int k0 = blockIdx.y * 32, n0 = blockIdx.x * 32;
    int r = threadIdx.x >> 5, c = threadIdx.x & 31;
    for (int rr = r; rr < 32; rr += 8)
        tile[rr][c] = src[eoff + (size_t)(k0 + rr) * N + n0 + c];
    __syncthreads();
    for (int rr = r; rr < 32; rr += 8)
        dst[eoff + (size_t)(n0 + rr) * K + k0 + c] = (f16)tile[c][rr];
}

// ---------------- LayerNorm: one block per token ----------------
__global__ __launch_bounds__(256) void layernorm_k(
    const float* __restrict__ x, const float* __restrict__ g, const float* __restrict__ b,
    f16* __restrict__ out_h, float* __restrict__ copy_f) {
    int t = blockIdx.x, tid = threadIdx.x;
    const float* row = x + (size_t)t * E_;
    float s = 0.f, s2 = 0.f;
    for (int i = tid; i < E_; i += 256) { float v = row[i]; s += v; s2 += v * v; }
    __shared__ float rs[256], rq[256];
    rs[tid] = s; rq[tid] = s2; __syncthreads();
    for (int st = 128; st > 0; st >>= 1) {
        if (tid < st) { rs[tid] += rs[tid + st]; rq[tid] += rq[tid + st]; }
        __syncthreads();
    }
    float mean = rs[0] * (1.f / E_);
    float var  = rq[0] * (1.f / E_) - mean * mean;
    float rstd = rsqrtf(var + 1e-5f);
    for (int i = tid; i < E_; i += 256) {
        float v = row[i];
        out_h[(size_t)t * E_ + i] = (f16)((v - mean) * rstd * g[i] + b[i]);
        if (copy_f) copy_f[(size_t)t * E_ + i] = v;   // seed residual / d_out
    }
}

// ---------------- TDM: async 2-D fp16 tile load into LDS ----------------
// Loads a (tileRows x tileCols) fp16 tile with row stride ldRow (elements),
// starting at gptr, into LDS at lds_byte_off. 2-group D# (2-D tensor).
__device__ __forceinline__ void tdm_load_tile_f16(
    const f16* gptr, unsigned lds_byte_off,
    unsigned tileCols, unsigned tileRows,
    unsigned tensorDim0, unsigned tensorDim1, unsigned ldRow) {
    unsigned long long ga = (unsigned long long)(uintptr_t)gptr;
    u32x4 g0;
    g0[0] = 1u;                               // count=1, user descriptor
    g0[1] = lds_byte_off;                     // lds_addr (bytes)
    g0[2] = (unsigned)ga;                     // global_addr[31:0]
    g0[3] = (unsigned)((ga >> 32) & 0x01FFFFFFu) | (2u << 30); // addr[56:32] | type=2
    u32x8 g1;
    g1[0] = (1u << 16);                       // data_size=1 (2 bytes)
    g1[1] = (tensorDim0 & 0xFFFFu) << 16;     // tensor_dim0[15:0]
    g1[2] = (tensorDim0 >> 16) | ((tensorDim1 & 0xFFFFu) << 16);
    g1[3] = (tensorDim1 >> 16) | (tileCols << 16);   // tile_dim0
    g1[4] = tileRows;                         // tile_dim1 (tile_dim2 = 0)
    g1[5] = ldRow;                            // tensor_dim0_stride[31:0]
    g1[6] = 0u;
    g1[7] = 0u;
    asm volatile("tensor_load_to_lds %0, %1" :: "s"(g0), "s"(g1) : "memory");
}

// ---------------- 128x128x32 WMMA GEMM, TDM-fed, double-buffered ----------------
// A: MxK fp16 row-major (lda). Bt: NxK fp16 row-major (ldb) — weights pre-transposed.
// Both 128x32 tiles are loaded by the Tensor Data Mover; waves only do
// ds_load_b128 -> v_wmma_f32_16x16x32_f16.
__global__ __launch_bounds__(256) void gemm_wmma_f16(
    const f16* __restrict__ A, int lda,
    const f16* __restrict__ Bt, int ldb,
    const float* __restrict__ bias,
    int M, int N, int K,
    int epi, float* __restrict__ Cf, f16* __restrict__ Ch,
    const float* __restrict__ resid,
    const float* __restrict__ combine, int ne_idx,
    float qscale, int qcols) {
    __shared__ f16 As[2][128 * 32];   // [row][k]
    __shared__ f16 Bs[2][128 * 32];   // [n][k]
    __shared__ int anyflag;

    const int tid  = threadIdx.x;
    const int lane = tid & 31;
    const int wave = tid >> 5;
    const int hi   = lane >> 4;
    const int ln   = lane & 15;
    const int tileM = blockIdx.y * 128;
    const int tileN = blockIdx.x * 128;

    // deterministic MoE tile-skip: no token in this 128-row tile uses expert ne_idx
    if (combine) {
        if (tid == 0) anyflag = 0;
        __syncthreads();
        for (int i = tid; i < 128; i += 256)
            if (combine[(size_t)(tileM + i) * NE_ + ne_idx] != 0.f) anyflag = 1;
        __syncthreads();
        if (!anyflag) return;
    }

    const int wm = wave & 1;   // 64-row half
    const int wn = wave >> 1;  // 32-col quarter

    v8f acc[4][2];
    for (int i = 0; i < 4; ++i)
        for (int j = 0; j < 2; ++j)
            for (int r = 0; r < 8; ++r) acc[i][j][r] = 0.f;

    // software pipeline: TDM pair for stage i+1 in flight while computing stage i
    if (wave == 0) {
        tdm_load_tile_f16(A  + (size_t)tileM * lda, (unsigned)(uintptr_t)&As[0][0],
                          32u, 128u, (unsigned)K, (unsigned)M, (unsigned)lda);
        tdm_load_tile_f16(Bt + (size_t)tileN * ldb, (unsigned)(uintptr_t)&Bs[0][0],
                          32u, 128u, (unsigned)K, (unsigned)N, (unsigned)ldb);
    }
    int buf = 0;
    for (int k0 = 0; k0 < K; k0 += 32, buf ^= 1) {
        const bool more = (k0 + 32 < K);
        if (wave == 0) {
            if (more) {
                tdm_load_tile_f16(A  + (size_t)tileM * lda + k0 + 32,
                                  (unsigned)(uintptr_t)&As[buf ^ 1][0],
                                  32u, 128u, (unsigned)K, (unsigned)M, (unsigned)lda);
                tdm_load_tile_f16(Bt + (size_t)tileN * ldb + k0 + 32,
                                  (unsigned)(uintptr_t)&Bs[buf ^ 1][0],
                                  32u, 128u, (unsigned)K, (unsigned)N, (unsigned)ldb);
                __builtin_amdgcn_s_wait_tensorcnt(2);   // in-order: current pair retired
            } else {
                __builtin_amdgcn_s_wait_tensorcnt(0);
            }
        }
        __syncthreads();   // publish TDM-written tiles to all waves

        // A lane layout k = hi*16 + e -> one contiguous v16h
        v16h af[4];
        for (int mf = 0; mf < 4; ++mf) {
            int m = wm * 64 + mf * 16 + ln;
            af[mf] = *(const v16h*)&As[buf][m * 32 + (hi << 4)];
        }
        // B lane layout k = hi*16 + e -> one contiguous v16h in Bs[n][k]
        v16h bf[2];
        for (int nf = 0; nf < 2; ++nf) {
            int n = wn * 32 + nf * 16 + ln;
            bf[nf] = *(const v16h*)&Bs[buf][n * 32 + (hi << 4)];
        }
        for (int mf = 0; mf < 4; ++mf)
            for (int nf = 0; nf < 2; ++nf)
                acc[mf][nf] = __builtin_amdgcn_wmma_f32_16x16x32_f16(
                    false, af[mf], false, bf[nf], (short)0, acc[mf][nf], false, false);
        __syncthreads();   // all reads of buf done before TDM refills it next round
    }

    // epilogue (C layout: m = r + 8*hi, n = ln)
    for (int mf = 0; mf < 4; ++mf)
        for (int nf = 0; nf < 2; ++nf)
            for (int r = 0; r < 8; ++r) {
                int row = tileM + wm * 64 + mf * 16 + r + (hi << 3);
                int col = tileN + wn * 32 + nf * 16 + ln;
                size_t idx = (size_t)row * N + col;
                float v = acc[mf][nf][r] + bias[col];
                if (epi == EPI_QKV) {
                    if (col < qcols) v *= qscale;
                    Ch[idx] = (f16)v;
                } else if (epi == EPI_GELU) {
                    Ch[idx] = (f16)(0.5f * v * (1.f + erff(v * 0.70710678118f)));
                } else if (epi == EPI_RESID) {
                    Cf[idx] = v + resid[idx];
                } else if (epi == EPI_COMBINE) {
                    Cf[idx] += combine[(size_t)row * NE_ + ne_idx] * v;
                } else {
                    Cf[idx] = v;
                }
            }
}

// ---------------- flash attention (per-(64 q rows, b, h) block, 4 waves) ----------------
__global__ __launch_bounds__(128) void attn_flash_k(
    const f16* __restrict__ qkv, f16* __restrict__ ctx) {
    __shared__ f16 Ks[32 * 64];      // [key][d]  (TDM-loaded, row-major)
    __shared__ f16 Vs[64 * 32];      // [d][key]  (manual transpose for contiguous B-frags)
    __shared__ f16 Pl[4][16 * 32];   // per-wave P scratch (C-layout -> A-layout)
    const int tid  = threadIdx.x;
    const int lane = tid & 31;
    const int wave = tid >> 5;
    const int hi   = lane >> 4;
    const int ln   = lane & 15;
    const int bh = blockIdx.y;
    const int b  = bh / H_;
    const int h  = bh % H_;
    const int q0 = blockIdx.x * 64 + wave * 16;

    // Q A-fragments (q pre-scaled by 1/sqrt(HD) in the QKV epilogue), resident all loop
    v16h qa[2];
    {
        const f16* qbase = qkv + ((size_t)((q0 + ln) * B_ + b)) * E3_ + h * HD_;
        for (int c = 0; c < 2; ++c) {
            qa[c] = *(const v16h*)&qbase[c * 32 + (hi << 4)];
        }
    }

    float Mr[8], Lr[8];
    v8f o[4];
    for (int r = 0; r < 8; ++r) { Mr[r] = -1e30f; Lr[r] = 0.f; }
    for (int f = 0; f < 4; ++f)
        for (int r = 0; r < 8; ++r) o[f][r] = 0.f;

    for (int kt = 0; kt < S_; kt += 32) {
        __syncthreads();   // Ks/Vs/Pl reusable
        // K tile (32 keys x 64 dims, key stride B*3E) via Tensor Data Mover
        if (wave == 0)
            tdm_load_tile_f16(qkv + ((size_t)(kt * B_ + b)) * E3_ + E_ + h * HD_,
                              (unsigned)(uintptr_t)&Ks[0],
                              64u, 32u, 64u, 32u, (unsigned)(B_ * E3_));
        // V tile transposed manually (TDM cannot transpose)
        for (int i = tid; i < (32 * 64) / 4; i += 128) {
            int key = i >> 4;
            int d4  = (i & 15) << 2;
            v4h vv = *(const v4h*)&qkv[(size_t)((kt + key) * B_ + b) * E3_ + 2 * E_ + h * HD_ + d4];
            Vs[(d4 + 0) * 32 + key] = vv[0];
            Vs[(d4 + 1) * 32 + key] = vv[1];
            Vs[(d4 + 2) * 32 + key] = vv[2];
            Vs[(d4 + 3) * 32 + key] = vv[3];
        }
        if (wave == 0) __builtin_amdgcn_s_wait_tensorcnt(0);
        __syncthreads();

        // scores: two 16x16 tiles (32 keys), HD=64 accumulated with 2 WMMAs each
        v8f s[2];
        for (int g = 0; g < 2; ++g) {
            v8f z;
            for (int r = 0; r < 8; ++r) z[r] = 0.f;
            int key = g * 16 + ln;
            v16h kb0 = *(const v16h*)&Ks[key * 64 + (hi << 4)];
            v16h kb1 = *(const v16h*)&Ks[key * 64 + 32 + (hi << 4)];
            z = __builtin_amdgcn_wmma_f32_16x16x32_f16(false, qa[0], false, kb0, (short)0, z, false, false);
            z = __builtin_amdgcn_wmma_f32_16x16x32_f16(false, qa[1], false, kb1, (short)0, z, false, false);
            s[g] = z;
        }
        // online softmax (row stats across the 16 lanes of each half-wave)
        float alpha[8];
        for (int r = 0; r < 8; ++r) {
            float mx = fmaxf(s[0][r], s[1][r]);
            for (int off = 1; off < 16; off <<= 1) mx = fmaxf(mx, __shfl_xor(mx, off, 16));
            float nm = fmaxf(Mr[r], mx);
            alpha[r] = __expf(Mr[r] - nm);
            Mr[r] = nm;
            float p0 = __expf(s[0][r] - nm);
            float p1 = __expf(s[1][r] - nm);
            s[0][r] = p0; s[1][r] = p1;
            float rsum = p0 + p1;
            for (int off = 1; off < 16; off <<= 1) rsum += __shfl_xor(rsum, off, 16);
            Lr[r] = Lr[r] * alpha[r] + rsum;
        }
        for (int f = 0; f < 4; ++f)
            for (int r = 0; r < 8; ++r) o[f][r] *= alpha[r];

        // C-layout P -> per-wave LDS -> A-layout fp16 fragment
        for (int r = 0; r < 8; ++r) {
            int m = r + (hi << 3);
            Pl[wave][m * 32 + ln]      = (f16)s[0][r];
            Pl[wave][m * 32 + 16 + ln] = (f16)s[1][r];
        }
        v16h pa;
        {
            pa = *(const v16h*)&Pl[wave][ln * 32 + (hi << 4)];
        }
        // o += P @ V  (V in [d][key] layout -> contiguous B-frags)
        for (int f = 0; f < 4; ++f) {
            v16h vb = *(const v16h*)&Vs[(f * 16 + ln) * 32 + (hi << 4)];
            o[f] = __builtin_amdgcn_wmma_f32_16x16x32_f16(false, pa, false, vb, (short)0, o[f], false, false);
        }
    }

    for (int f = 0; f < 4; ++f)
        for (int r = 0; r < 8; ++r) {
            int qs = q0 + r + (hi << 3);
            float val = o[f][r] / Lr[r];
            ctx[((size_t)(qs * B_ + b)) * E_ + h * HD_ + f * 16 + ln] = (f16)val;
        }
}

// ---------------- gate + top-2 + softmax-over-topk ----------------
__global__ __launch_bounds__(256) void gate_topk_k(
    const f16* __restrict__ xn, const float* __restrict__ gw, const float* __restrict__ gb,
    float* __restrict__ comb) {
    int t = blockIdx.x, tid = threadIdx.x;
    int e = tid >> 5, lane = tid & 31;
    __shared__ float lg[NE_];
    float p = 0.f;
    for (int i = lane; i < E_; i += 32)
        p += (float)xn[(size_t)t * E_ + i] * gw[(size_t)e * E_ + i];
    for (int off = 1; off < 32; off <<= 1) p += __shfl_xor(p, off, 32);
    if (lane == 0) lg[e] = p + gb[e];
    __syncthreads();
    if (tid == 0) {
        float b1v = -1e30f; int i1 = 0;
        for (int k = 0; k < NE_; ++k) if (lg[k] > b1v) { b1v = lg[k]; i1 = k; }
        float b2v = -1e30f; int i2 = 0;
        for (int k = 0; k < NE_; ++k) if (k != i1 && lg[k] > b2v) { b2v = lg[k]; i2 = k; }
        float g1 = 1.f / (1.f + __expf(b2v - b1v));   // softmax over the selected pair
        float g2 = 1.f - g1;
        for (int k = 0; k < NE_; ++k) comb[(size_t)t * NE_ + k] = 0.f;
        comb[(size_t)t * NE_ + i1] = g1;
        comb[(size_t)t * NE_ + i2] = g2;
    }
}

// ---------------- host-side orchestration ----------------
extern "C" void kernel_launch(void* const* d_in, const int* in_sizes, int n_in,
                              void* d_out, int out_size, void* d_ws, size_t ws_size,
                              hipStream_t stream) {
    (void)in_sizes; (void)n_in; (void)out_size; (void)ws_size;
    const float* x         = (const float*)d_in[0];
    const float* in_proj_w = (const float*)d_in[1];
    const float* in_proj_b = (const float*)d_in[2];
    const float* out_w     = (const float*)d_in[3];
    const float* out_b     = (const float*)d_in[4];
    const float* ln1_g     = (const float*)d_in[5];
    const float* ln1_b     = (const float*)d_in[6];
    const float* ln2_g     = (const float*)d_in[7];
    const float* ln2_b     = (const float*)d_in[8];
    const float* gate_w    = (const float*)d_in[9];
    const float* gate_b    = (const float*)d_in[10];
    const float* w1        = (const float*)d_in[11];
    const float* b1        = (const float*)d_in[12];
    const float* w2        = (const float*)d_in[13];
    const float* b2        = (const float*)d_in[14];
    float* out = (float*)d_out;

    // carve workspace
    size_t off = 0;
    char* base = (char*)d_ws;
    auto carve = [&](size_t bytes) -> void* {
        void* p = base + off;
        off += (bytes + 255) & ~(size_t)255;
        return p;
    };
    f16*   xn1  = (f16*)carve((size_t)T_ * E_ * 2);
    f16*   qkv  = (f16*)carve((size_t)T_ * E3_ * 2);
    f16*   ctx  = (f16*)carve((size_t)T_ * E_ * 2);
    float* x1   = (float*)carve((size_t)T_ * E_ * 4);
    f16*   xn2  = (f16*)carve((size_t)T_ * E_ * 2);
    float* comb = (float*)carve((size_t)T_ * NE_ * 4);
    f16*   hbuf = (f16*)carve((size_t)T_ * FF_ * 2);
    f16*   wqkh = (f16*)carve((size_t)E3_ * E_ * 2);     // [3E,E]  (already N,K)
    f16*   woth = (f16*)carve((size_t)E_ * E_ * 2);      // [E,E]   (already N,K)
    f16*   w1t  = (f16*)carve((size_t)NE_ * FF_ * E_ * 2);  // [e][FF,E]  transposed
    f16*   w2t  = (f16*)carve((size_t)NE_ * E_ * FF_ * 2);  // [e][E,FF]  transposed

    // 1) weight conversion (one-time, bandwidth-trivial); MoE weights transposed to [N,K]
    cvt_f32_to_f16<<<4096, 256, 0, stream>>>(in_proj_w, wqkh, (size_t)E3_ * E_);
    cvt_f32_to_f16<<<4096, 256, 0, stream>>>(out_w, woth, (size_t)E_ * E_);
    cvt_tr_f32_to_f16<<<dim3(FF_ / 32, E_ / 32, NE_), 256, 0, stream>>>(w1, w1t, E_, FF_);
    cvt_tr_f32_to_f16<<<dim3(E_ / 32, FF_ / 32, NE_), 256, 0, stream>>>(w2, w2t, FF_, E_);

    // 2) LN1
    layernorm_k<<<T_, 256, 0, stream>>>(x, ln1_g, ln1_b, xn1, nullptr);

    // 3) QKV projection, fused bias + q-scale, fp16 out
    gemm_wmma_f16<<<dim3(E3_ / 128, T_ / 128), 256, 0, stream>>>(
        xn1, E_, wqkh, E_, in_proj_b, T_, E3_, E_,
        EPI_QKV, nullptr, qkv, nullptr, nullptr, 0, 0.125f, E_);

    // 4) flash attention
    attn_flash_k<<<dim3(S_ / 64, B_ * H_), 128, 0, stream>>>(qkv, ctx);

    // 5) output projection + residual -> x1 (fp32)
    gemm_wmma_f16<<<dim3(E_ / 128, T_ / 128), 256, 0, stream>>>(
        ctx, E_, woth, E_, out_b, T_, E_, E_,
        EPI_RESID, x1, nullptr, x, nullptr, 0, 1.f, 0);

    // 6) LN2 (also seeds d_out with residual x1)
    layernorm_k<<<T_, 256, 0, stream>>>(x1, ln2_g, ln2_b, xn2, out);

    // 7) gate: top-2 + softmax over selected
    gate_topk_k<<<T_, 256, 0, stream>>>(xn2, gate_w, gate_b, comb);

    // 8) MoE experts: h = gelu(xn2 @ w1[e] + b1[e]); out += combine[:,e] * (h @ w2[e] + b2[e])
    for (int e = 0; e < NE_; ++e) {
        gemm_wmma_f16<<<dim3(FF_ / 128, T_ / 128), 256, 0, stream>>>(
            xn2, E_, w1t + (size_t)e * FF_ * E_, E_, b1 + (size_t)e * FF_,
            T_, FF_, E_, EPI_GELU, nullptr, hbuf, nullptr, comb, e, 1.f, 0);
        gemm_wmma_f16<<<dim3(E_ / 128, T_ / 128), 256, 0, stream>>>(
            hbuf, FF_, w2t + (size_t)e * E_ * FF_, FF_, b2 + (size_t)e * E_,
            T_, E_, FF_, EPI_COMBINE, out, nullptr, nullptr, comb, e, 1.f, 0);
    }
}
